// MoEMLP_7937099563380
// MI455X (gfx1250) — matrix operand order repaired and checked by reading, not compiled, not run on hardware
//
#include <hip/hip_runtime.h>
#include <hip/hip_bf16.h>

typedef __bf16 bf16;
typedef __attribute__((ext_vector_type(8)))  __bf16 v8bf;
typedef __attribute__((ext_vector_type(16))) __bf16 v16bf;
typedef __attribute__((ext_vector_type(8)))  float  v8f;

constexpr int T = 4096;   // B*S tokens
constexpr int D = 1024;   // model dim
constexpr int E = 8;      // experts
constexpr int W = 1024;   // expert width

// workspace layout (bytes)
constexpr size_t WS_CNT  = 0;                            // E ints
constexpr size_t WS_OFF  = 256;                          // E+1 ints
constexpr size_t WS_TOK  = 512;                          // E*T ints   (t*2+slot)
constexpr size_t WS_GATE = WS_TOK  + (size_t)E * T * 4;  // E*T floats
constexpr size_t WS_A    = WS_GATE + (size_t)E * T * 4;  // 2T*W bf16  (relu^2 out)
constexpr size_t WS_CTR  = WS_A    + (size_t)2 * T * W * 2; // 2T*D floats

union Frag16 { v16bf v; v8bf h[2]; };

static __device__ __forceinline__ v8bf cvt8(float4 a, float4 b) {
    v8bf r;
    r[0]=(bf16)a.x; r[1]=(bf16)a.y; r[2]=(bf16)a.z; r[3]=(bf16)a.w;
    r[4]=(bf16)b.x; r[5]=(bf16)b.y; r[6]=(bf16)b.z; r[7]=(bf16)b.w;
    return r;
}

// ---------------------------------------------------------------- init
__global__ void moe_init(int* __restrict__ cnt) {
    if (threadIdx.x < E) cnt[threadIdx.x] = 0;
}

// ---------------------------------------------------------------- router
__global__ void moe_router(const float* __restrict__ x,
                           const float* __restrict__ rw_g,
                           int* __restrict__ cnt,
                           int* __restrict__ tok,
                           float* __restrict__ gate) {
    __shared__ float rw[E][D];                     // 32 KB
    const int tid = threadIdx.x;
    for (int i = tid; i < E * D; i += 256)
        rw[i >> 10][i & (D - 1)] = rw_g[i];
    __syncthreads();

    const int t = blockIdx.x * 256 + tid;
    float acc[E];
#pragma unroll
    for (int e = 0; e < E; ++e) acc[e] = 0.f;

    const float* xr = x + (size_t)t * D;
    for (int d = 0; d < D; d += 4) {
        float4 xv = *(const float4*)(xr + d);
#pragma unroll
        for (int e = 0; e < E; ++e) {
            acc[e] = fmaf(xv.x, rw[e][d + 0],
                     fmaf(xv.y, rw[e][d + 1],
                     fmaf(xv.z, rw[e][d + 2],
                     fmaf(xv.w, rw[e][d + 3], acc[e]))));
        }
    }
    float p[E];
#pragma unroll
    for (int e = 0; e < E; ++e) p[e] = 1.f / (1.f + expf(-acc[e]));

    int i0 = 0; float p0 = p[0];
#pragma unroll
    for (int e = 1; e < E; ++e) if (p[e] > p0) { p0 = p[e]; i0 = e; }
    int i1 = -1; float p1 = -1e30f;
#pragma unroll
    for (int e = 0; e < E; ++e) if (e != i0 && p[e] > p1) { p1 = p[e]; i1 = e; }

    const float s = 1.f / (p0 + p1 + 1e-20f);
    int pos0 = atomicAdd(&cnt[i0], 1);
    tok[i0 * T + pos0]  = t * 2 + 0;
    gate[i0 * T + pos0] = p0 * s;
    int pos1 = atomicAdd(&cnt[i1], 1);
    tok[i1 * T + pos1]  = t * 2 + 1;
    gate[i1 * T + pos1] = p1 * s;
}

// ---------------------------------------------------------------- scan
__global__ void moe_scan(const int* __restrict__ cnt, int* __restrict__ off) {
    if (threadIdx.x == 0) {
        int a = 0;
        for (int e = 0; e < E; ++e) { off[e] = a; a += cnt[e]; }
        off[E] = a;
    }
}

// ---------------------------------------------------------------- up-proj
// Grid: (W/128, T/128, E). Block: 256 threads (8 wave32s). C tile = 128x128.
// Double-buffered LDS, 8 WMMA per wave per k-step.
__global__ void __launch_bounds__(256)
moe_up(const float* __restrict__ x, const float* __restrict__ w1,
       const int* __restrict__ cnt, const int* __restrict__ off,
       const int* __restrict__ tok, bf16* __restrict__ abuf) {
    const int e   = blockIdx.z;
    const int nct = cnt[e];
    const int mt  = blockIdx.y;
    if (mt * 128 >= nct) return;
    const int abase = off[e];
    const int nb0   = blockIdx.x * 128;

    __shared__ __align__(16) bf16 As[2][128][40];
    __shared__ __align__(16) bf16 Bs[2][128][40];
    __shared__ int tokS[128];

    const int tid = threadIdx.x;
    if (tid < 128) {
        int g = mt * 128 + tid;
        tokS[tid] = (g < nct) ? (tok[e * T + g] >> 1) : 0;
    }
    __syncthreads();

    // --- staging maps (2 units per thread each for A and B) ---
    const int ar0 = tid >> 2,        aoct0 = (tid & 3) * 8;   // rows 0..63
    const int ar1 = (tid + 256) >> 2;                          // rows 64..127
    const int aoct1 = aoct0;
    const int bn  = tid & 127;                                 // col 0..127
    const int bq0 = (tid >> 7) * 8;                            // k-octet 0/8
    const int bq1 = bq0 + 16;                                  // k-octet 16/24

    const float* aS0 = x + (size_t)tokS[ar0] * D + aoct0;      // hoisted gather ptrs
    const float* aS1 = x + (size_t)tokS[ar1] * D + aoct1;
    const float* bS  = w1 + (size_t)e * W + nb0 + bn;          // + (kk+q+k)*E*W

    auto stageA = [&](int buf, int kk) {
        float4 f0 = *(const float4*)(aS0 + kk);
        float4 f1 = *(const float4*)(aS0 + kk + 4);
        *(v8bf*)&As[buf][ar0][aoct0] = cvt8(f0, f1);
        float4 g0 = *(const float4*)(aS1 + kk);
        float4 g1 = *(const float4*)(aS1 + kk + 4);
        *(v8bf*)&As[buf][ar1][aoct1] = cvt8(g0, g1);
    };
    auto stageB = [&](int buf, int kk) {
        const float* p0 = bS + (size_t)(kk + bq0) * (E * W);
        const float* p1 = bS + (size_t)(kk + bq1) * (E * W);
        float a[8], b[8];
#pragma unroll
        for (int k = 0; k < 8; ++k) a[k] = p0[(size_t)k * (E * W)];
#pragma unroll
        for (int k = 0; k < 8; ++k) b[k] = p1[(size_t)k * (E * W)];
        v8bf va, vb;
#pragma unroll
        for (int k = 0; k < 8; ++k) { va[k] = (bf16)a[k]; vb[k] = (bf16)b[k]; }
        *(v8bf*)&Bs[buf][bn][bq0] = va;
        *(v8bf*)&Bs[buf][bn][bq1] = vb;
    };

    const int lane = tid & 31, wm = tid >> 5;     // 8 waves x 16 rows
    const int nl   = lane & 15;
    const int akb  = (lane >> 4) * 8;
    const int bkb  = (lane >> 4) * 8;
    v8f acc[8];
#pragma unroll
    for (int j = 0; j < 8; ++j) acc[j] = (v8f){};

    stageA(0, 0);
    stageB(0, 0);
    __syncthreads();

    constexpr int KSTEPS = D / 32;
    for (int s = 0; s < KSTEPS; ++s) {
        const int cur = s & 1, nxt = cur ^ 1;
        if (s + 1 < KSTEPS) {
            stageA(nxt, (s + 1) * 32);
            stageB(nxt, (s + 1) * 32);
            __builtin_prefetch(bS + (size_t)((s + 2) * 32) * (E * W), 0, 0);
        }
        const int am = wm * 16 + nl;
        Frag16 af;
        af.h[0] = *(const v8bf*)&As[cur][am][akb];
        af.h[1] = *(const v8bf*)&As[cur][am][akb + 16];
#pragma unroll
        for (int j = 0; j < 8; ++j) {
            Frag16 bfr;
            bfr.h[0] = *(const v8bf*)&Bs[cur][j * 16 + nl][bkb];
            bfr.h[1] = *(const v8bf*)&Bs[cur][j * 16 + nl][bkb + 16];
            acc[j] = __builtin_amdgcn_wmma_f32_16x16x32_bf16(
                        false, af.v, false, bfr.v, (short)0, acc[j], false, false);
        }
        __syncthreads();
    }

    // relu^2, store bf16 intermediate
    const int mloc = wm * 16 + ((lane >> 4) ? 8 : 0);
#pragma unroll
    for (int r = 0; r < 8; ++r) {
        const int g = mt * 128 + mloc + r;
        if (g < nct) {
            const size_t row = (size_t)(abase + g) * W + nb0;
#pragma unroll
            for (int j = 0; j < 8; ++j) {
                float h = acc[j][r];
                h = h > 0.f ? h * h : 0.f;
                abuf[row + j * 16 + nl] = (bf16)h;
            }
        }
    }
}

// ---------------------------------------------------------------- down-proj
// Grid: (D/128, T/128, E). A operand already bf16 -> async copy to LDS.
__global__ void __launch_bounds__(256)
moe_down(const bf16* __restrict__ abuf, const float* __restrict__ w2,
         const int* __restrict__ cnt, const int* __restrict__ off,
         const int* __restrict__ tok, const float* __restrict__ gate,
         float* __restrict__ contrib) {
    const int e   = blockIdx.z;
    const int nct = cnt[e];
    const int mt  = blockIdx.y;
    if (mt * 128 >= nct) return;
    const int abase = off[e];
    const int nb0   = blockIdx.x * 128;

    __shared__ __align__(16) bf16 As[2][128][40];
    __shared__ __align__(16) bf16 Bs[2][128][40];
    __shared__ int   tokS[128];
    __shared__ float gateS[128];

    const int tid = threadIdx.x;
    if (tid < 128) {
        int g = mt * 128 + tid;
        bool ok = g < nct;
        tokS[tid]  = ok ? tok[e * T + g]  : 0;
        gateS[tid] = ok ? gate[e * T + g] : 0.f;
    }

    const int ar0 = tid >> 2,        aoct0 = (tid & 3) * 8;
    const int ar1 = (tid + 256) >> 2;
    const int bn  = tid & 127;
    const int bq0 = (tid >> 7) * 8;
    const int bq1 = bq0 + 16;

    const int g0 = mt * 128 + ar0, g1 = mt * 128 + ar1;
    const int as0 = (g0 < nct) ? g0 : 0;
    const int as1 = (g1 < nct) ? g1 : 0;
    const bf16* aS0 = abuf + (size_t)(abase + as0) * W + aoct0;
    const bf16* aS1 = abuf + (size_t)(abase + as1) * W + aoct0;
    const float* bS = w2 + (size_t)e * W * D + nb0 + bn;       // + (kk+q+k)*D

    // async copy of the (already bf16) A tile: global -> LDS, ASYNCcnt tracked
    auto stageA_async = [&](int buf, int kk) {
        unsigned d0 = (unsigned)(uintptr_t)&As[buf][ar0][aoct0];
        unsigned d1 = (unsigned)(uintptr_t)&As[buf][ar1][aoct0];
        const bf16* s0 = aS0 + kk;
        const bf16* s1 = aS1 + kk;
        asm volatile("global_load_async_to_lds_b128 %0, %1, off"
                     :: "v"(d0), "v"(s0) : "memory");
        asm volatile("global_load_async_to_lds_b128 %0, %1, off"
                     :: "v"(d1), "v"(s1) : "memory");
    };
    auto stageB = [&](int buf, int kk) {
        const float* p0 = bS + (size_t)(kk + bq0) * D;
        const float* p1 = bS + (size_t)(kk + bq1) * D;
        float a[8], b[8];
#pragma unroll
        for (int k = 0; k < 8; ++k) a[k] = p0[(size_t)k * D];
#pragma unroll
        for (int k = 0; k < 8; ++k) b[k] = p1[(size_t)k * D];
        v8bf va, vb;
#pragma unroll
        for (int k = 0; k < 8; ++k) { va[k] = (bf16)a[k]; vb[k] = (bf16)b[k]; }
        *(v8bf*)&Bs[buf][bn][bq0] = va;
        *(v8bf*)&Bs[buf][bn][bq1] = vb;
    };

    const int lane = tid & 31, wm = tid >> 5;
    const int nl   = lane & 15;
    const int akb  = (lane >> 4) * 8;
    const int bkb  = (lane >> 4) * 8;
    v8f acc[8];
#pragma unroll
    for (int j = 0; j < 8; ++j) acc[j] = (v8f){};

    stageA_async(0, 0);
    stageB(0, 0);
    asm volatile("s_wait_asynccnt 0x0" ::: "memory");
    __syncthreads();

    constexpr int KSTEPS = W / 32;
    for (int s = 0; s < KSTEPS; ++s) {
        const int cur = s & 1, nxt = cur ^ 1;
        const bool more = (s + 1 < KSTEPS);
        if (more) {
            stageA_async(nxt, (s + 1) * 32);
            stageB(nxt, (s + 1) * 32);
            __builtin_prefetch(bS + (size_t)((s + 2) * 32) * D, 0, 0);
        }
        const int am = wm * 16 + nl;
        Frag16 af;
        af.h[0] = *(const v8bf*)&As[cur][am][akb];
        af.h[1] = *(const v8bf*)&As[cur][am][akb + 16];
#pragma unroll
        for (int j = 0; j < 8; ++j) {
            Frag16 bfr;
            bfr.h[0] = *(const v8bf*)&Bs[cur][j * 16 + nl][bkb];
            bfr.h[1] = *(const v8bf*)&Bs[cur][j * 16 + nl][bkb + 16];
            acc[j] = __builtin_amdgcn_wmma_f32_16x16x32_bf16(
                        false, af.v, false, bfr.v, (short)0, acc[j], false, false);
        }
        if (more) asm volatile("s_wait_asynccnt 0x0" ::: "memory");
        __syncthreads();
    }

    const int mloc = wm * 16 + ((lane >> 4) ? 8 : 0);
#pragma unroll
    for (int r = 0; r < 8; ++r) {
        const int g = mt * 128 + mloc + r;
        if (g < nct) {
            const float gt = gateS[mloc + r];
            const size_t row = (size_t)tokS[mloc + r] * D + nb0;
#pragma unroll
            for (int j = 0; j < 8; ++j)
                contrib[row + j * 16 + nl] = acc[j][r] * gt;
        }
    }
}

// ---------------------------------------------------------------- combine
__global__ void moe_combine(const float* __restrict__ contrib, float* __restrict__ out) {
    const int i  = blockIdx.x * 256 + threadIdx.x;  // over T*(D/4)
    const int t  = i >> 8;                          // D/4 == 256
    const int dq = i & 255;
    const float4* c4 = (const float4*)contrib;
    float4 a = c4[(size_t)(2 * t)     * 256 + dq];
    float4 b = c4[(size_t)(2 * t + 1) * 256 + dq];
    float4 r; r.x = a.x + b.x; r.y = a.y + b.y; r.z = a.z + b.z; r.w = a.w + b.w;
    ((float4*)out)[i] = r;
}

// ---------------------------------------------------------------- launch
extern "C" void kernel_launch(void* const* d_in, const int* in_sizes, int n_in,
                              void* d_out, int out_size, void* d_ws, size_t ws_size,
                              hipStream_t stream) {
    const float* x  = (const float*)d_in[0];
    const float* rw = (const float*)d_in[1];
    const float* w1 = (const float*)d_in[2];
    const float* w2 = (const float*)d_in[3];
    float* out = (float*)d_out;

    char*  ws   = (char*)d_ws;
    int*   cnt  = (int*)(ws + WS_CNT);
    int*   off  = (int*)(ws + WS_OFF);
    int*   tok  = (int*)(ws + WS_TOK);
    float* gate = (float*)(ws + WS_GATE);
    bf16*  abuf = (bf16*)(ws + WS_A);
    float* ctr  = (float*)(ws + WS_CTR);

    hipLaunchKernelGGL(moe_init,   dim3(1),       dim3(32),  0, stream, cnt);
    hipLaunchKernelGGL(moe_router, dim3(T / 256), dim3(256), 0, stream, x, rw, cnt, tok, gate);
    hipLaunchKernelGGL(moe_scan,   dim3(1),       dim3(1),   0, stream, cnt, off);
    hipLaunchKernelGGL(moe_up,     dim3(W / 128, T / 128, E), dim3(256), 0, stream,
                       x, w1, cnt, off, tok, abuf);
    hipLaunchKernelGGL(moe_down,   dim3(D / 128, T / 128, E), dim3(256), 0, stream,
                       abuf, w2, cnt, off, tok, gate, ctr);
    hipLaunchKernelGGL(moe_combine, dim3(T * (D / 4) / 256), dim3(256), 0, stream, ctr, out);
}
